// PolicyNet_44942537786008
// MI455X (gfx1250) — matrix operand order repaired and checked by reading, not compiled, not run on hardware
//
#include <hip/hip_runtime.h>

// ---------------------------------------------------------------------------
// Problem constants: S=11, F=650 -> SF=7150 (pad 7168), H=1024, A=47, B=4096
// ---------------------------------------------------------------------------
#define BB   4096
#define SF   7150
#define KP0  7168          // SF rounded up to multiple of 32
#define HH   1024
#define AA   47
#define NEGV (-1.0e9f)

typedef __bf16 bf16_t;
typedef __attribute__((ext_vector_type(8)))  __bf16 v8bf;
typedef __attribute__((ext_vector_type(16))) __bf16 v16bf;
typedef __attribute__((ext_vector_type(8)))  float  v8f;

// fp32 -> (hi, lo) bf16 split: f ~= hi + lo, relative error ~2^-16
__device__ __forceinline__ void cvt_split(float f, bf16_t& hi, bf16_t& lo) {
    bf16_t h = (bf16_t)f;           // RNE
    hi = h;
    lo = (bf16_t)(f - (float)h);
}

__device__ __forceinline__ v8f wmma_bf16(v16bf a, v16bf b, v8f c) {
    return __builtin_amdgcn_wmma_f32_16x16x32_bf16(
        false, a, false, b, (short)0, c, false, false);
}

__device__ __forceinline__ v16bf cat8(v8bf a, v8bf b) {
    return __builtin_shufflevector(a, b, 0, 1, 2, 3, 4, 5, 6, 7,
                                         8, 9, 10, 11, 12, 13, 14, 15);
}

// ---------------------------------------------------------------------------
// Prep 1: split fp32 rows into bf16 hi/lo planes, zero-padding K -> Kp.
// One block per row; fully coalesced.
// ---------------------------------------------------------------------------
__global__ __launch_bounds__(256) void split_rows_kernel(
        const float* __restrict__ X, bf16_t* __restrict__ Xhi,
        bf16_t* __restrict__ Xlo, int K, int Kp) {
    int m = blockIdx.x;
    const float* row = X + (size_t)m * K;
    bf16_t* oh = Xhi + (size_t)m * Kp;
    bf16_t* ol = Xlo + (size_t)m * Kp;
    for (int k = threadIdx.x; k < Kp; k += 256) {
        float f = (k < K) ? row[k] : 0.0f;
        bf16_t h, l;
        cvt_split(f, h, l);
        oh[k] = h;
        ol[k] = l;
    }
}

// ---------------------------------------------------------------------------
// Prep 2: W (KxN, row-major f32) -> transposed bf16 hi/lo planes (NxKp),
// zero-padded K. 32x32 LDS tile transpose; coalesced both directions.
// ---------------------------------------------------------------------------
__global__ __launch_bounds__(256) void transpose_split_kernel(
        const float* __restrict__ W, bf16_t* __restrict__ Thi,
        bf16_t* __restrict__ Tlo, int K, int N, int Kp) {
    __shared__ float tile[32][33];
    int tx = threadIdx.x & 31;
    int ty = threadIdx.x >> 5;        // 0..7
    int k0 = blockIdx.y * 32;
    int n0 = blockIdx.x * 32;
#pragma unroll
    for (int r = ty; r < 32; r += 8) {
        int k = k0 + r, n = n0 + tx;
        tile[r][tx] = (k < K && n < N) ? W[(size_t)k * N + n] : 0.0f;
    }
    __syncthreads();
#pragma unroll
    for (int r = ty; r < 32; r += 8) {
        int n = n0 + r, k = k0 + tx;
        if (n < N && k < Kp) {
            bf16_t h, l;
            cvt_split(tile[tx][r], h, l);
            Thi[(size_t)n * Kp + k] = h;
            Tlo[(size_t)n * Kp + k] = l;
        }
    }
}

// ---------------------------------------------------------------------------
// GEMM: C(MxN) = A(MxKp) @ B^T(NxKp) + bias, bf16x3 planes, fp32 accumulate.
// A planes row-major MxKp; B planes are transposed weights, row-major NxKp.
// Block: 256 threads = 8 waves (4x2); wave tile 32x64 (2x4 WMMA accs);
// block tile 128x128. Kp % 32 == 0, M % 128 == 0, N % 128 == 0.
// Inner loop: 24 x global_load_b128 + 24 x v_wmma per K-chunk, zero VALU cvt.
// ---------------------------------------------------------------------------
__global__ __launch_bounds__(256) void gemm_bf16_planes_kernel(
        const bf16_t* __restrict__ Ahi, const bf16_t* __restrict__ Alo,
        const bf16_t* __restrict__ Bhi, const bf16_t* __restrict__ Blo,
        const float* __restrict__ bias, float* __restrict__ C,
        int M, int Kp, int N) {
    int lane = threadIdx.x & 31;
    int wave = threadIdx.x >> 5;
    int wrow = wave >> 1;             // 0..3
    int wcol = wave & 1;              // 0..1
    int m0 = blockIdx.y * 128 + wrow * 32;
    int n0 = blockIdx.x * 128 + wcol * 64;
    int half = lane >> 4;
    int l15 = lane & 15;

    v8f zero = {0.f, 0.f, 0.f, 0.f, 0.f, 0.f, 0.f, 0.f};
    v8f acc[2][4];
#pragma unroll
    for (int i = 0; i < 2; ++i)
#pragma unroll
        for (int j = 0; j < 4; ++j) acc[i][j] = zero;

    // Per-lane row bases (elements).
    size_t aro[2], bro[4];
#pragma unroll
    for (int i = 0; i < 2; ++i) aro[i] = (size_t)(m0 + 16 * i + l15) * Kp;
#pragma unroll
    for (int j = 0; j < 4; ++j) bro[j] = (size_t)(n0 + 16 * j + l15) * Kp;

    for (int k0 = 0; k0 < Kp; k0 += 32) {
        // A frag: lane half h covers K runs [k0+8h .. +7] and [k0+16+8h .. +7]
        int kb0 = k0 + 8 * half;
        int kb1 = k0 + 16 + 8 * half;
        // B frag: lane half h covers K runs [k0+8h .. +7] and [k0+16+8h .. +7]
        int kbB = k0 + 8 * half;

        v16bf ah[2], al[2];
#pragma unroll
        for (int i = 0; i < 2; ++i) {
            ah[i] = cat8(*(const v8bf*)(Ahi + aro[i] + kb0),
                         *(const v8bf*)(Ahi + aro[i] + kb1));
            al[i] = cat8(*(const v8bf*)(Alo + aro[i] + kb0),
                         *(const v8bf*)(Alo + aro[i] + kb1));
        }
        v16bf bh[4], bl[4];
#pragma unroll
        for (int j = 0; j < 4; ++j) {
            bh[j] = cat8(*(const v8bf*)(Bhi + bro[j] + kbB),
                         *(const v8bf*)(Bhi + bro[j] + kbB + 16));
            bl[j] = cat8(*(const v8bf*)(Blo + bro[j] + kbB),
                         *(const v8bf*)(Blo + bro[j] + kbB + 16));
        }
#pragma unroll
        for (int i = 0; i < 2; ++i) {
#pragma unroll
            for (int j = 0; j < 4; ++j) {
                acc[i][j] = wmma_bf16(ah[i], bh[j], acc[i][j]);  // hi*hi
                acc[i][j] = wmma_bf16(ah[i], bl[j], acc[i][j]);  // hi*lo
                acc[i][j] = wmma_bf16(al[i], bh[j], acc[i][j]);  // lo*hi
            }
        }
    }

    // 32-bit C/D 16x16 layout: VGPR r -> M = r + 8*half, N = l15.
#pragma unroll
    for (int i = 0; i < 2; ++i) {
#pragma unroll
        for (int j = 0; j < 4; ++j) {
            int nn = n0 + 16 * j + l15;
            float bv = bias[nn];
#pragma unroll
            for (int r = 0; r < 8; ++r) {
                int mm = m0 + 16 * i + r + 8 * half;
                C[(size_t)mm * N + nn] = acc[i][j][r] + bv;
            }
        }
    }
}

// ---------------------------------------------------------------------------
// In-place LayerNorm + ReLU over rows of length 1024. One block per row.
// ---------------------------------------------------------------------------
__global__ __launch_bounds__(256) void ln_relu_kernel(
        float* __restrict__ X, const float* __restrict__ g,
        const float* __restrict__ be) {
    int b = blockIdx.x;
    int tid = threadIdx.x;
    float* row = X + (size_t)b * HH;
    float v[4];
#pragma unroll
    for (int i = 0; i < 4; ++i) v[i] = row[tid + i * 256];

    __shared__ float red[256];
    red[tid] = v[0] + v[1] + v[2] + v[3];
    __syncthreads();
    for (int off = 128; off > 0; off >>= 1) {
        if (tid < off) red[tid] += red[tid + off];
        __syncthreads();
    }
    float mean = red[0] * (1.0f / HH);
    __syncthreads();

    float vs = 0.f;
#pragma unroll
    for (int i = 0; i < 4; ++i) { float d = v[i] - mean; vs += d * d; }
    red[tid] = vs;
    __syncthreads();
    for (int off = 128; off > 0; off >>= 1) {
        if (tid < off) red[tid] += red[tid + off];
        __syncthreads();
    }
    float inv = rsqrtf(red[0] * (1.0f / HH) + 1e-5f);
#pragma unroll
    for (int i = 0; i < 4; ++i) {
        int c = tid + i * 256;
        float y = (v[i] - mean) * inv * g[c] + be[c];
        row[c] = fmaxf(y, 0.0f);
    }
}

// ---------------------------------------------------------------------------
// Policy + value head (95 cols, 0.8 GFLOP): x row staged in LDS,
// Wp reads coalesced per-k; memory-bound, L2-resident weights.
// ---------------------------------------------------------------------------
__global__ __launch_bounds__(128) void head_kernel(
        const float* __restrict__ X, const float* __restrict__ Wp,
        const float* __restrict__ bp, const float* __restrict__ Wv,
        const float* __restrict__ bvv, float* __restrict__ logits,
        float* __restrict__ value) {
    __shared__ float xs[HH];
    int b = blockIdx.x, tid = threadIdx.x;
    const float* row = X + (size_t)b * HH;
    for (int i = tid; i < HH; i += 128) xs[i] = row[i];
    __syncthreads();

    if (tid < 2 * AA) {
        float acc = 0.f;
#pragma unroll 8
        for (int k = 0; k < HH; ++k) acc += xs[k] * Wp[(size_t)k * (2 * AA) + tid];
        logits[(size_t)b * (2 * AA) + tid] = acc + bp[tid];
    } else if (tid == 2 * AA) {
        float acc = 0.f;
#pragma unroll 8
        for (int k = 0; k < HH; ++k) acc += xs[k] * Wv[k];
        value[b] = acc + bvv[0];
    }
}

// ---------------------------------------------------------------------------
// Per-row masking / Gumbel argmax / log-softmax (exact reference semantics).
// ---------------------------------------------------------------------------
__global__ __launch_bounds__(64) void sample_kernel(
        const float* __restrict__ logits, const int* __restrict__ amask,
        const float* __restrict__ gumbel, float* __restrict__ logp_out,
        float* __restrict__ act_out) {
    int b = blockIdx.x, tid = threadIdx.x;
    __shared__ float s_l[AA];
    __shared__ float s_sc[AA];
    __shared__ int   s_m2[AA];
    __shared__ int   s_a1;
    __shared__ float s_lp1;

    size_t base = (size_t)b * (2 * AA);

    if (tid < AA) {
        float lg = logits[base + tid];
        float l1 = (amask[base + tid] == 0) ? NEGV : lg;
        s_l[tid]  = l1;
        s_sc[tid] = l1 + gumbel[base + tid];
    }
    __syncthreads();
    if (tid == 0) {
        int a1 = 0; float best = s_sc[0];
        for (int i = 1; i < AA; ++i) if (s_sc[i] > best) { best = s_sc[i]; a1 = i; }
        float mx = s_l[0];
        for (int i = 1; i < AA; ++i) mx = fmaxf(mx, s_l[i]);
        float se = 0.f;
        for (int i = 0; i < AA; ++i) se += expf(s_l[i] - mx);
        s_lp1 = s_l[a1] - mx - logf(se);
        s_a1 = a1;
    }
    __syncthreads();
    int a1 = s_a1;

    if (tid < AA) {
        int m2 = amask[base + AA + tid];
        if (a1 >= 1 && a1 <= 6 && tid == a1) m2 = 0;
        if (a1 > 26 && a1 <= 46 && tid >= 27 && tid < AA) m2 = 0;
        if (a1 == 0 && tid == 0) m2 = 0;
        s_m2[tid] = m2;
    }
    __syncthreads();
    if (tid == 0) {
        int ssum = 0;
        for (int i = 0; i < AA; ++i) ssum += s_m2[i];
        if (ssum == 0 && a1 == 0) s_m2[0] = 1;
    }
    __syncthreads();

    if (tid < AA) {
        float lg = logits[base + AA + tid];
        float l2 = (s_m2[tid] == 0) ? NEGV : lg;
        s_l[tid]  = l2;
        s_sc[tid] = l2 + gumbel[base + AA + tid];
    }
    __syncthreads();
    if (tid == 0) {
        int a2 = 0; float best = s_sc[0];
        for (int i = 1; i < AA; ++i) if (s_sc[i] > best) { best = s_sc[i]; a2 = i; }
        float mx = s_l[0];
        for (int i = 1; i < AA; ++i) mx = fmaxf(mx, s_l[i]);
        float se = 0.f;
        for (int i = 0; i < AA; ++i) se += expf(s_l[i] - mx);
        float lp2 = s_l[a2] - mx - logf(se);
        logp_out[b * 2 + 0] = s_lp1;
        logp_out[b * 2 + 1] = lp2;
        act_out[b * 2 + 0] = (float)a1;
        act_out[b * 2 + 1] = (float)a2;
    }
}

// ---------------------------------------------------------------------------
extern "C" void kernel_launch(void* const* d_in, const int* in_sizes, int n_in,
                              void* d_out, int out_size, void* d_ws, size_t ws_size,
                              hipStream_t stream) {
    const float* obs   = (const float*)d_in[0];
    const int*   amask = (const int*)  d_in[1];
    const float* gumb  = (const float*)d_in[2];
    const float* W0 = (const float*)d_in[3];
    const float* b0 = (const float*)d_in[4];
    const float* g0 = (const float*)d_in[5];
    const float* be0= (const float*)d_in[6];
    const float* W1 = (const float*)d_in[7];
    const float* b1 = (const float*)d_in[8];
    const float* g1 = (const float*)d_in[9];
    const float* be1= (const float*)d_in[10];
    const float* W2 = (const float*)d_in[11];
    const float* b2 = (const float*)d_in[12];
    const float* g2 = (const float*)d_in[13];
    const float* be2= (const float*)d_in[14];
    const float* Wp = (const float*)d_in[15];
    const float* bp = (const float*)d_in[16];
    const float* Wv = (const float*)d_in[17];
    const float* bv = (const float*)d_in[18];

    // ---- workspace carve-up (256B aligned), total ~189 MB ----
    size_t off = 0;
    char* base = (char*)d_ws;
    auto alloc = [&](size_t bytes) -> char* {
        char* p = base + off;
        off += (bytes + 255) & ~(size_t)255;
        return p;
    };
    bf16_t* X0hi  = (bf16_t*)alloc((size_t)BB * KP0 * 2);   // 58.7 MB
    bf16_t* X0lo  = (bf16_t*)alloc((size_t)BB * KP0 * 2);   // 58.7 MB
    bf16_t* W0thi = (bf16_t*)alloc((size_t)HH * KP0 * 2);   // 14.7 MB
    bf16_t* W0tlo = (bf16_t*)alloc((size_t)HH * KP0 * 2);   // 14.7 MB
    bf16_t* W1thi = (bf16_t*)alloc((size_t)HH * HH * 2);    // 2.1 MB x4
    bf16_t* W1tlo = (bf16_t*)alloc((size_t)HH * HH * 2);
    bf16_t* W2thi = (bf16_t*)alloc((size_t)HH * HH * 2);
    bf16_t* W2tlo = (bf16_t*)alloc((size_t)HH * HH * 2);
    float*  bufF  = (float*) alloc((size_t)BB * HH * 4);    // 16.8 MB
    bf16_t* Hhi   = (bf16_t*)alloc((size_t)BB * HH * 2);    // 8.4 MB x2
    bf16_t* Hlo   = (bf16_t*)alloc((size_t)BB * HH * 2);

    // ---- output layout: logits (B,2,A) | logp (B,2) | actions (B,2) | value (B)
    float* out    = (float*)d_out;
    float* logits = out;
    float* logp   = logits + (size_t)BB * 2 * AA;
    float* acts   = logp   + (size_t)BB * 2;
    float* val    = acts   + (size_t)BB * 2;

    dim3 gemm_grid(HH / 128, BB / 128);   // 8 x 32

    // Prep: one-pass fp32 -> bf16 hi/lo planes (bandwidth-bound)
    split_rows_kernel<<<BB, 256, 0, stream>>>(obs, X0hi, X0lo, SF, KP0);
    transpose_split_kernel<<<dim3(HH / 32, KP0 / 32), 256, 0, stream>>>(
        W0, W0thi, W0tlo, SF, HH, KP0);
    transpose_split_kernel<<<dim3(HH / 32, HH / 32), 256, 0, stream>>>(
        W1, W1thi, W1tlo, HH, HH, HH);
    transpose_split_kernel<<<dim3(HH / 32, HH / 32), 256, 0, stream>>>(
        W2, W2thi, W2tlo, HH, HH, HH);

    // Layer 0
    gemm_bf16_planes_kernel<<<gemm_grid, 256, 0, stream>>>(
        X0hi, X0lo, W0thi, W0tlo, b0, bufF, BB, KP0, HH);
    ln_relu_kernel<<<BB, 256, 0, stream>>>(bufF, g0, be0);

    // Layer 1
    split_rows_kernel<<<BB, 256, 0, stream>>>(bufF, Hhi, Hlo, HH, HH);
    gemm_bf16_planes_kernel<<<gemm_grid, 256, 0, stream>>>(
        Hhi, Hlo, W1thi, W1tlo, b1, bufF, BB, HH, HH);
    ln_relu_kernel<<<BB, 256, 0, stream>>>(bufF, g1, be1);

    // Layer 2
    split_rows_kernel<<<BB, 256, 0, stream>>>(bufF, Hhi, Hlo, HH, HH);
    gemm_bf16_planes_kernel<<<gemm_grid, 256, 0, stream>>>(
        Hhi, Hlo, W2thi, W2tlo, b2, bufF, BB, HH, HH);
    ln_relu_kernel<<<BB, 256, 0, stream>>>(bufF, g2, be2);

    // Heads + sampling
    head_kernel<<<BB, 128, 0, stream>>>(bufF, Wp, bp, Wv, bv, logits, val);
    sample_kernel<<<BB, 64, 0, stream>>>(logits, amask, gumb, logp, acts);
}
